// HybridGNNv5_19765439496860
// MI455X (gfx1250) — compile-verified, matrix operand order checked by reading, not run on hardware
//
#include <hip/hip_runtime.h>
#include <hip/hip_bf16.h>

#define EPSV 1e-5f

typedef __attribute__((ext_vector_type(16))) _Float16 v16h;
typedef __attribute__((ext_vector_type(8)))  _Float16 v8h;
typedef __attribute__((ext_vector_type(4)))  _Float16 v4h;
typedef __attribute__((ext_vector_type(8)))  float    v8f;

// ---------------------------------------------------------------- utilities
__global__ void k_fill(float* __restrict__ p, float v, long n) {
    long gid = (long)blockIdx.x * blockDim.x + threadIdx.x;
    if (gid < n) p[gid] = v;
}

__device__ inline void atomicMaxF(float* addr, float val) {
    int* ai = (int*)addr;
    int old = __float_as_int(*addr);
    while (__int_as_float(old) < val) {
        int assumed = old;
        old = atomicCAS(ai, assumed, __float_as_int(val));
        if (old == assumed) break;
    }
}

// ------------------------------------------------- WMMA GEMM: Y = A@W + bias
// A: [Nrows, K] f32, W: [K, 128] f32, Y: [Nrows, 128] f32.
// 256 threads (8 waves); 128 rows per block; wave w owns rows w*16..w*16+15.
// f16 operands staged in LDS; W is pre-swizzled into fragment-ready order so
// every fragment (A and B) is exactly two ds_load_b128 reads.
// Accumulator starts as WMMA inline-0 (C folded to constant); bias is added
// in the store epilogue. Accumulation in f32 via v_wmma_f32_16x16x32_f16.

template<int KK, bool FULL>
__device__ __forceinline__
void wmma_tiles(const _Float16* __restrict__ sWf, const v16h (&afrag)[KK],
                const float (&bv)[8], float* __restrict__ ypBase,
                int rowsLeft, int lane)
{
#pragma unroll
    for (int t = 0; t < 8; ++t) {
        v8f acc = {};                          // folds to inline 0 in first WMMA
#pragma unroll
        for (int kk = 0; kk < KK; ++kk) {
            const _Float16* wp = sWf + (((kk * 8 + t) * 32) + lane) * 16;
            v8h b0 = *(const v8h*)wp;
            v8h b1 = *(const v8h*)(wp + 8);
            v16h bfrag = __builtin_shufflevector(b0, b1,
                0, 1, 2, 3, 4, 5, 6, 7, 8, 9, 10, 11, 12, 13, 14, 15);
            acc = __builtin_amdgcn_wmma_f32_16x16x32_f16(
                false, afrag[kk], false, bfrag, (short)0, acc, false, false);
        }
        float* yp = ypBase + t * 16;
#pragma unroll
        for (int r = 0; r < 8; ++r)
            if (FULL || r < rowsLeft) yp[r * 128] = acc[r] + bv[t]; // imm-offset stores
    }
}

template<int K>
__global__ __launch_bounds__(256)
void k_gemm_wmma(const float* __restrict__ A, const float* __restrict__ W,
                 const float* __restrict__ bias, float* __restrict__ Y, int Nrows)
{
    constexpr int KK = K / 32;                 // k-steps
    __shared__ _Float16 sWf[K * 128];          // swizzled: [kk][t][lane][16]
    __shared__ _Float16 sA[8][16 * K];         // per-wave row-major f16 tiles
    const int tid  = threadIdx.x;
    const int w    = tid >> 5;
    const int lane = tid & 31;
    const int g    = lane >> 4;   // half-wave group
    const int nlo  = lane & 15;   // M for A-frag, N for B-frag

    // preload bias (overlaps with LDS staging below)
    float bv[8];
#pragma unroll
    for (int t = 0; t < 8; ++t) bv[t] = bias[t * 16 + nlo];

    // ---- stage W, swizzled into per-lane fragment order ----
    // dest d = ((kk*8 + t)*32 + lane)*16 + i
    // elem i of a fragment lives at K-index kidx(i,g) (ISA 7.12.2 16-bit map)
    for (int d = tid; d < K * 128; d += 256) {
        int i  = d & 15;
        int ln = (d >> 4) & 31;
        int t  = (d >> 9) & 7;
        int kk = d >> 12;
        int gg = ln >> 4, nl = ln & 15;
        int v = i >> 1, lo = i & 1;
        int k = kk * 32 + (v < 4 ? 2 * v : 16 + 2 * (v - 4)) + lo + 8 * gg;
        sWf[d] = (_Float16)W[k * 128 + t * 16 + nl];
    }

    // ---- stage A tile (128 rows x K), float4 loads -> 4-half LDS stores ----
    const int rowBase = blockIdx.x * 128;      // fits int: N*128 < 2^31
    constexpr int KQ = K / 4;
    for (int i = tid; i < 128 * KQ; i += 256) {
        int r = i / KQ, kq = i - r * KQ;
        int row = rowBase + r;
        float4 v = make_float4(0.f, 0.f, 0.f, 0.f);
        if (row < Nrows) v = *(const float4*)&A[row * K + kq * 4];
        v4h hv = { (_Float16)v.x, (_Float16)v.y, (_Float16)v.z, (_Float16)v.w };
        *(v4h*)&sA[r >> 4][(r & 15) * K + kq * 4] = hv;
    }
    __syncthreads();

    // ---- preload A fragments: two b128 LDS reads each ----
    v16h afrag[KK];
#pragma unroll
    for (int kk = 0; kk < KK; ++kk) {
        const _Float16* ap = &sA[w][nlo * K + kk * 32 + 8 * g];
        v8h a0 = *(const v8h*)ap;          // K = base+0..7   (VGPRs 0-3)
        v8h a1 = *(const v8h*)(ap + 16);   // K = base+16..23 (VGPRs 4-7)
        afrag[kk] = __builtin_shufflevector(a0, a1,
            0, 1, 2, 3, 4, 5, 6, 7, 8, 9, 10, 11, 12, 13, 14, 15);
    }

    const int row0 = rowBase + w * 16 + g * 8;
    float* ypBase = Y + row0 * 128 + nlo;      // 32-bit offset, one 64-bit add
    if (rowBase + 128 <= Nrows)
        wmma_tiles<KK, true >(sWf, afrag, bv, ypBase, 8, lane);
    else
        wmma_tiles<KK, false>(sWf, afrag, bv, ypBase, Nrows - row0, lane);
}

// --------------------------------------------- column stats for BatchNorm
__global__ __launch_bounds__(256)
void k_colstats(const float* __restrict__ Y, int Nrows,
                float* __restrict__ csum, float* __restrict__ csq)
{
    __shared__ float ssum[256], ssq[256];
    const int c = threadIdx.x & 127;
    const int half = threadIdx.x >> 7;
    float s = 0.f, q = 0.f;
    int r0 = blockIdx.x * 64;
    int r1 = min(r0 + 64, Nrows);
    for (int r = r0 + half; r < r1; r += 2) {
        float v = Y[(long)r * 128 + c];
        s += v; q += v * v;
    }
    ssum[threadIdx.x] = s; ssq[threadIdx.x] = q;
    __syncthreads();
    if (half == 0) {
        atomicAdd(&csum[c], ssum[c] + ssum[c + 128]);
        atomicAdd(&csq[c],  ssq[c]  + ssq[c + 128]);
    }
}

__global__ void k_bn_relu(float* __restrict__ Y, const float* __restrict__ csum,
                          const float* __restrict__ csq, const float* __restrict__ gam,
                          const float* __restrict__ bet, long n, float invN)
{
    long gid = (long)blockIdx.x * blockDim.x + threadIdx.x;
    if (gid >= n) return;
    int c = gid & 127;
    float mu = csum[c] * invN;
    float var = csq[c] * invN - mu * mu;
    float v = Y[gid];
    Y[gid] = fmaxf(gam[c] * (v - mu) * rsqrtf(var + EPSV) + bet[c], 0.0f);
}

// --------------------------- GINE message: msg = relu(h[src]+relu(ea@We+be))
__global__ __launch_bounds__(256)
void k_message(const float* __restrict__ h, const float* __restrict__ edge_attr,
               const float* __restrict__ w_e, const float* __restrict__ b_e,
               const int* __restrict__ src, const int* __restrict__ dst,
               float* __restrict__ agg, int E)
{
    __shared__ float swe[8 * 128];
    __shared__ float sbe[128];
    __shared__ float ea[2][8];
    const int tid = threadIdx.x;
    for (int i = tid; i < 8 * 128; i += 256) swe[i] = w_e[i];
    if (tid < 128) sbe[tid] = b_e[tid];
    if (tid < 16) {
        int jj = blockIdx.x * 2 + (tid >> 3);
        ea[tid >> 3][tid & 7] = (jj < E) ? edge_attr[(long)jj * 8 + (tid & 7)] : 0.0f;
    }
    __syncthreads();
    const int sub = tid >> 7, c = tid & 127;
    const int j = blockIdx.x * 2 + sub;
    if (j >= E) return;
    float acc = sbe[c];
#pragma unroll
    for (int k = 0; k < 8; ++k) acc += ea[sub][k] * swe[k * 128 + c];
    float ev = fmaxf(acc, 0.0f);
    float m = fmaxf(h[(long)src[j] * 128 + c] + ev, 0.0f);
    atomicAdd(&agg[(long)dst[j] * 128 + c], m);
}

// z = (1+eps_l)*h + agg  (in-place on agg)
__global__ void k_z(const float* __restrict__ hin, float* __restrict__ agg,
                    const float* __restrict__ eps, int l, long n)
{
    long gid = (long)blockIdx.x * blockDim.x + threadIdx.x;
    if (gid >= n) return;
    agg[gid] = (1.0f + eps[l]) * hin[gid] + agg[gid];
}

// out = relu(LN(m2) + idn)   -- one wave32 per row
__global__ __launch_bounds__(256)
void k_ln_res(const float* __restrict__ m2, const float* __restrict__ idn,
              const float* __restrict__ lng, const float* __restrict__ lnb,
              float* __restrict__ outl, int Nrows)
{
    int row = blockIdx.x * 8 + (threadIdx.x >> 5);
    int lane = threadIdx.x & 31;
    if (row >= Nrows) return;
    float v[4]; float s = 0.f, q = 0.f;
#pragma unroll
    for (int t = 0; t < 4; ++t) {
        v[t] = m2[(long)row * 128 + lane + t * 32];
        s += v[t]; q += v[t] * v[t];
    }
#pragma unroll
    for (int off = 16; off; off >>= 1) {
        s += __shfl_xor(s, off, 32);
        q += __shfl_xor(q, off, 32);
    }
    float mu = s * (1.0f / 128.0f);
    float var = q * (1.0f / 128.0f) - mu * mu;
    float rs = rsqrtf(var + EPSV);
#pragma unroll
    for (int t = 0; t < 4; ++t) {
        int c = lane + t * 32;
        float ln = lng[c] * (v[t] - mu) * rs + lnb[c];
        outl[(long)row * 128 + c] = fmaxf(ln + idn[(long)row * 128 + c], 0.0f);
    }
}

// JK attention scores: scores[i,l] = relu(out_il @ w1 + b1) @ w2 + b2
__global__ __launch_bounds__(256)
void k_jk_scores(const float* __restrict__ outs, int Nrows,
                 const float* __restrict__ w1, const float* __restrict__ b1,
                 const float* __restrict__ w2, const float* __restrict__ b2,
                 float* __restrict__ scores)
{
    __shared__ float sW1[128 * 64];
    __shared__ float sB1[64], sW2[64];
    const int tid = threadIdx.x;
    for (int i = tid; i < 128 * 64; i += 256) sW1[i] = w1[i];
    if (tid < 64) { sB1[tid] = b1[tid]; sW2[tid] = w2[tid]; }
    __syncthreads();
    const int w = tid >> 5, lane = tid & 31;
    const float b2v = b2[0];
    const long total = (long)Nrows * 3;
    for (long item = (long)blockIdx.x * 8 + w; item < total; item += (long)gridDim.x * 8) {
        long i = item / 3; int l = (int)(item - i * 3);
        const float* vec = outs + ((long)l * Nrows + i) * 128;
        float a0 = sB1[lane], a1 = sB1[lane + 32];
        for (int k = 0; k < 128; ++k) {
            float x = vec[k];                  // broadcast load
            a0 += x * sW1[k * 64 + lane];
            a1 += x * sW1[k * 64 + lane + 32];
        }
        float p = fmaxf(a0, 0.f) * sW2[lane] + fmaxf(a1, 0.f) * sW2[lane + 32];
#pragma unroll
        for (int off = 16; off; off >>= 1) p += __shfl_xor(p, off, 32);
        if (lane == 0) scores[item] = p + b2v;
    }
}

// softmax over L=3, xn = sum_l attn*outs
__global__ void k_attn_xn(const float* __restrict__ outs, const float* __restrict__ scores,
                          float* __restrict__ xn, int Nrows)
{
    long gid = (long)blockIdx.x * blockDim.x + threadIdx.x;
    if (gid >= (long)Nrows * 128) return;
    long i = gid >> 7; int c = gid & 127;
    float s0 = scores[i * 3 + 0], s1 = scores[i * 3 + 1], s2 = scores[i * 3 + 2];
    float m = fmaxf(s0, fmaxf(s1, s2));
    float e0 = expf(s0 - m), e1 = expf(s1 - m), e2 = expf(s2 - m);
    float inv = 1.0f / (e0 + e1 + e2);
    long NH = (long)Nrows * 128;
    float v = e0 * outs[i * 128 + c] + e1 * outs[NH + i * 128 + c] + e2 * outs[2 * NH + i * 128 + c];
    xn[gid] = v * inv;
}

// segment sum/count/max pooling over batch
__global__ void k_pool(const float* __restrict__ xn, const int* __restrict__ batch,
                       float* __restrict__ s, float* __restrict__ cnt,
                       float* __restrict__ mx, int Nrows)
{
    long gid = (long)blockIdx.x * blockDim.x + threadIdx.x;
    if (gid >= (long)Nrows * 128) return;
    long i = gid >> 7; int c = gid & 127;
    int b = batch[i];
    float v = xn[gid];
    atomicAdd(&s[b * 128 + c], v);
    atomicMaxF(&mx[b * 128 + c], v);
    if (c == 0) atomicAdd(&cnt[b], 1.0f);
}

// gf = graph_features @ gf_w + gf_b   [G,10]@[10,32]
__global__ void k_gf(const float* __restrict__ gfeat, const float* __restrict__ w,
                     const float* __restrict__ b, float* __restrict__ out, int G)
{
    int gid = blockIdx.x * blockDim.x + threadIdx.x;
    if (gid >= G * 32) return;
    int g = gid >> 5, u = gid & 31;
    float acc = b[u];
#pragma unroll
    for (int k = 0; k < 10; ++k) acc += gfeat[g * 10 + k] * w[k * 32 + u];
    out[gid] = acc;
}

// te = relu(LN(tfidf @ tf_w + tf_b))  [G,TF]@[TF,128]; block per graph
__global__ __launch_bounds__(128)
void k_te(const float* __restrict__ tfidf, const float* __restrict__ w,
          const float* __restrict__ b, const float* __restrict__ lng,
          const float* __restrict__ lnb, float* __restrict__ out, int TF)
{
    __shared__ float rs[128], rq[128];
    const int g = blockIdx.x, c = threadIdx.x;
    float acc = b[c];
    const float* tv = tfidf + (long)g * TF;
    for (int k = 0; k < TF; ++k) acc += tv[k] * w[(long)k * 128 + c];
    rs[c] = acc; rq[c] = acc * acc;
    __syncthreads();
    for (int off = 64; off; off >>= 1) {
        if (c < off) { rs[c] += rs[c + off]; rq[c] += rq[c + off]; }
        __syncthreads();
    }
    float mu = rs[0] * (1.0f / 128.0f);
    float var = rq[0] * (1.0f / 128.0f) - mu * mu;
    out[g * 128 + c] = fmaxf(lng[c] * (acc - mu) * rsqrtf(var + EPSV) + lnb[c], 0.0f);
}

// comb = LN(concat[mean, max, sum, gf, te])  per-graph, 544 features
__global__ __launch_bounds__(256)
void k_comb(const float* __restrict__ s, const float* __restrict__ cnt,
            const float* __restrict__ mx, const float* __restrict__ gf,
            const float* __restrict__ te, const float* __restrict__ lng,
            const float* __restrict__ lnb, float* __restrict__ comb)
{
    __shared__ float sh[544];
    __shared__ float rs[256], rq[256];
    const int g = blockIdx.x, tid = threadIdx.x;
    float ls = 0.f, lq = 0.f;
    for (int c = tid; c < 544; c += 256) {
        float v;
        if (c < 128)      v = s[g * 128 + c] / fmaxf(cnt[g], 1.0f);
        else if (c < 256) v = mx[g * 128 + (c - 128)];
        else if (c < 384) v = s[g * 128 + (c - 256)];
        else if (c < 416) v = gf[g * 32 + (c - 384)];
        else              v = te[g * 128 + (c - 416)];
        sh[c] = v; ls += v; lq += v * v;
    }
    rs[tid] = ls; rq[tid] = lq;
    __syncthreads();
    for (int off = 128; off; off >>= 1) {
        if (tid < off) { rs[tid] += rs[tid + off]; rq[tid] += rq[tid + off]; }
        __syncthreads();
    }
    float mu = rs[0] * (1.0f / 544.0f);
    float var = rq[0] * (1.0f / 544.0f) - mu * mu;
    float r = rsqrtf(var + EPSV);
    for (int c = tid; c < 544; c += 256)
        comb[g * 544 + c] = lng[c] * (sh[c] - mu) * r + lnb[c];
}

// y = comb @ cls_w1 + cls_b1   [G,544]@[544,128]
__global__ __launch_bounds__(128)
void k_cls1(const float* __restrict__ comb, const float* __restrict__ w,
            const float* __restrict__ b, float* __restrict__ y)
{
    const int g = blockIdx.x, c = threadIdx.x;
    float acc = b[c];
    const float* cv = comb + g * 544;
    for (int k = 0; k < 544; ++k) acc += cv[k] * w[k * 128 + c];
    y[g * 128 + c] = acc;
}

__global__ __launch_bounds__(128)
void k_clsstats(const float* __restrict__ y, float* __restrict__ mu,
                float* __restrict__ var, int G)
{
    const int c = threadIdx.x;
    float s = 0.f, q = 0.f;
    for (int r = 0; r < G; ++r) { float v = y[r * 128 + c]; s += v; q += v * v; }
    float m = s / (float)G;
    mu[c] = m;
    var[c] = q / (float)G - m * m;
}

__global__ __launch_bounds__(128)
void k_final(const float* __restrict__ y, const float* __restrict__ mu,
             const float* __restrict__ var, const float* __restrict__ bg,
             const float* __restrict__ bb, const float* __restrict__ w2,
             const float* __restrict__ b2, float* __restrict__ out)
{
    __shared__ float red[128];
    const int g = blockIdx.x, c = threadIdx.x;
    float v = y[g * 128 + c];
    v = bg[c] * (v - mu[c]) * rsqrtf(var[c] + EPSV) + bb[c];
    v = fmaxf(v, 0.0f) * w2[c];
    red[c] = v;
    __syncthreads();
    for (int off = 64; off; off >>= 1) {
        if (c < off) red[c] += red[c + off];
        __syncthreads();
    }
    if (c == 0) out[g] = red[0] + b2[0];
}

// ================================================================ launcher
extern "C" void kernel_launch(void* const* d_in, const int* in_sizes, int n_in,
                              void* d_out, int out_size, void* d_ws, size_t ws_size,
                              hipStream_t stream)
{
    const float* x        = (const float*)d_in[0];
    const float* e_attr   = (const float*)d_in[1];
    const float* gfeat    = (const float*)d_in[2];
    const float* tfidf    = (const float*)d_in[3];
    const float* w_in     = (const float*)d_in[4];
    const float* b_in     = (const float*)d_in[5];
    const float* g_in     = (const float*)d_in[6];
    const float* be_in    = (const float*)d_in[7];
    const float* w_e      = (const float*)d_in[8];
    const float* b_e      = (const float*)d_in[9];
    const float* conv_w1  = (const float*)d_in[10];
    const float* conv_b1  = (const float*)d_in[11];
    const float* conv_bng = (const float*)d_in[12];
    const float* conv_bnb = (const float*)d_in[13];
    const float* conv_w2  = (const float*)d_in[14];
    const float* conv_b2  = (const float*)d_in[15];
    const float* conv_eps = (const float*)d_in[16];
    const float* conv_lng = (const float*)d_in[17];
    const float* conv_lnb = (const float*)d_in[18];
    const float* jk_w1    = (const float*)d_in[19];
    const float* jk_b1    = (const float*)d_in[20];
    const float* jk_w2    = (const float*)d_in[21];
    const float* jk_b2    = (const float*)d_in[22];
    const float* gf_w     = (const float*)d_in[23];
    const float* gf_b     = (const float*)d_in[24];
    const float* tf_w     = (const float*)d_in[25];
    const float* tf_b     = (const float*)d_in[26];
    const float* tf_lng   = (const float*)d_in[27];
    const float* tf_lnb   = (const float*)d_in[28];
    const float* pre_lng  = (const float*)d_in[29];
    const float* pre_lnb  = (const float*)d_in[30];
    const float* cls_w1   = (const float*)d_in[31];
    const float* cls_b1   = (const float*)d_in[32];
    const float* cls_bng  = (const float*)d_in[33];
    const float* cls_bnb  = (const float*)d_in[34];
    const float* cls_w2   = (const float*)d_in[35];
    const float* cls_b2   = (const float*)d_in[36];
    const int*   e_index  = (const int*)d_in[37];
    const int*   batch    = (const int*)d_in[38];

    const int N  = in_sizes[38];
    const int E  = in_sizes[1] / 8;
    const int G  = in_sizes[2] / 10;
    const int TF = in_sizes[3] / G;
    const long NH = (long)N * 128;

    const int* src = e_index;
    const int* dst = e_index + E;

    // workspace layout (all f32)
    float* ws     = (float*)d_ws;
    float* h0     = ws;                 // N*128
    float* outs   = h0 + NH;            // 3*N*128 ([L,N,H])
    float* agg    = outs + 3 * NH;      // N*128 (agg / z / m2 / xn)
    float* tmp    = agg + NH;           // N*128 (m1)
    float* colsum = tmp + NH;           // 128
    float* colsq  = colsum + 128;       // 128
    float* scores = colsq + 128;        // N*3
    float* spool  = scores + (long)N * 3;   // G*128
    float* cntp   = spool + (long)G * 128;  // G
    float* mxp    = cntp + G;               // G*128
    float* gfb    = mxp + (long)G * 128;    // G*32
    float* teb    = gfb + (long)G * 32;     // G*128
    float* combb  = teb + (long)G * 128;    // G*544
    float* ybuf   = combb + (long)G * 544;  // G*128
    float* clsmu  = ybuf + (long)G * 128;   // 128
    float* clsvar = clsmu + 128;            // 128

    const float invN = 1.0f / (float)N;
    dim3 blk(256);
    const int ewBlocks = (int)((NH + 255) / 256);

    // ---- input projection: h0 = relu(BN(x @ w_in + b_in)) ----
    k_fill<<<1, 256, 0, stream>>>(colsum, 0.f, 256);
    k_gemm_wmma<64><<<(N + 127) / 128, blk, 0, stream>>>(x, w_in, b_in, h0, N);
    k_colstats<<<(N + 63) / 64, blk, 0, stream>>>(h0, N, colsum, colsq);
    k_bn_relu<<<ewBlocks, blk, 0, stream>>>(h0, colsum, colsq, g_in, be_in, NH, invN);

    // ---- 3 GINE layers ----
    const float* hin = h0;
    for (int l = 0; l < 3; ++l) {
        float* outl = outs + (long)l * NH;
        k_fill<<<ewBlocks, blk, 0, stream>>>(agg, 0.f, NH);
        k_message<<<(E + 1) / 2, blk, 0, stream>>>(hin, e_attr, w_e, b_e, src, dst, agg, E);
        k_z<<<ewBlocks, blk, 0, stream>>>(hin, agg, conv_eps, l, NH);
        // m1 = relu(BN(z @ w1 + b1))
        k_fill<<<1, 256, 0, stream>>>(colsum, 0.f, 256);
        k_gemm_wmma<128><<<(N + 127) / 128, blk, 0, stream>>>(
            agg, conv_w1 + (long)l * 128 * 128, conv_b1 + l * 128, tmp, N);
        k_colstats<<<(N + 63) / 64, blk, 0, stream>>>(tmp, N, colsum, colsq);
        k_bn_relu<<<ewBlocks, blk, 0, stream>>>(tmp, colsum, colsq,
                                                conv_bng + l * 128, conv_bnb + l * 128, NH, invN);
        // m2 = m1 @ w2 + b2 (into agg, z dead)
        k_gemm_wmma<128><<<(N + 127) / 128, blk, 0, stream>>>(
            tmp, conv_w2 + (long)l * 128 * 128, conv_b2 + l * 128, agg, N);
        // h = relu(LN(m2) + hin)
        k_ln_res<<<(N + 7) / 8, blk, 0, stream>>>(agg, hin, conv_lng + l * 128,
                                                  conv_lnb + l * 128, outl, N);
        hin = outl;
    }

    // ---- jumping-knowledge attention ----
    k_jk_scores<<<960, blk, 0, stream>>>(outs, N, jk_w1, jk_b1, jk_w2, jk_b2, scores);
    k_attn_xn<<<ewBlocks, blk, 0, stream>>>(outs, scores, agg /*xn*/, N);

    // ---- segment pooling ----
    k_fill<<<(G * 128 + 255) / 256, blk, 0, stream>>>(spool, 0.f, (long)G * 128);
    k_fill<<<1, 256, 0, stream>>>(cntp, 0.f, G);
    k_fill<<<(G * 128 + 255) / 256, blk, 0, stream>>>(mxp, -1e30f, (long)G * 128);
    k_pool<<<ewBlocks, blk, 0, stream>>>(agg, batch, spool, cntp, mxp, N);

    // ---- graph head ----
    k_gf<<<(G * 32 + 255) / 256, blk, 0, stream>>>(gfeat, gf_w, gf_b, gfb, G);
    k_te<<<G, 128, 0, stream>>>(tfidf, tf_w, tf_b, tf_lng, tf_lnb, teb, TF);
    k_comb<<<G, 256, 0, stream>>>(spool, cntp, mxp, gfb, teb, pre_lng, pre_lnb, combb);
    k_cls1<<<G, 128, 0, stream>>>(combb, cls_w1, cls_b1, ybuf);
    k_clsstats<<<1, 128, 0, stream>>>(ybuf, clsmu, clsvar, G);
    k_final<<<G, 128, 0, stream>>>(ybuf, clsmu, clsvar, cls_bng, cls_bnb,
                                   cls_w2, cls_b2, (float*)d_out);
}
